// LSTMModel_79353815761341
// MI455X (gfx1250) — matrix operand order repaired and checked by reading, not compiled, not run on hardware
//
#include <hip/hip_runtime.h>

// ---------------------------------------------------------------------------
// Fused 2-layer LSTM (linear cell activation) + dense head for MI455X gfx1250.
// B=64, T=2048, D=128, H=256, 4H=1024, O=64.
// 4 independent workgroups (one per 16-row batch block), each a full WGP
// (1024 threads = 32 wave32s). Per timestep z = [x_t|h] @ [W;U] + b via
// v_wmma_f32_16x16x32_bf16; gates combined through an LDS exchange buffer;
// cell state c stays fp32 in registers; h is bf16 in LDS. Weights pre-packed
// N-major bf16 (~1.9MB, L2-resident). Weight pointers are laundered per
// timestep to block LICM (prevents scratch spills) and the loads go through
// explicit address_space(1) pointers so they stay global_load_b128 (LOADcnt
// only -- no flat/DScnt entanglement). x is prefetched 2 steps ahead
// (global_prefetch_b8) to keep HBM latency off the serial chain.
// ---------------------------------------------------------------------------

typedef __bf16          v16bf  __attribute__((ext_vector_type(16)));
typedef float           v8f    __attribute__((ext_vector_type(8)));
typedef unsigned short  u16x8  __attribute__((ext_vector_type(8)));
typedef unsigned short  u16x16 __attribute__((ext_vector_type(16)));

// explicit global-address-space vector pointer (forces global_load encoding)
typedef __attribute__((address_space(1))) const u16x16* gbl_u16x16_p;

#define T_    2048
#define D_    128
#define H_    256
#define NG_   1024          // 4*H
#define K1_   384           // D + H   (layer-1 concatenated K)
#define K2_   512           // H + H   (layer-2 concatenated K)
#define O_    64

#define XA_STRIDE 136       // 128 cols + pad (bank-conflict mitigation)
#define H_STRIDE  264       // 256 cols + pad
#define SMEM_BYTES (65536 + (16*XA_STRIDE + 2*16*H_STRIDE) * 2)

__device__ __forceinline__ unsigned short f2bf(float f) {
    unsigned u = __builtin_bit_cast(unsigned, f);
    u += 0x7FFFu + ((u >> 16) & 1u);          // round-to-nearest-even
    return (unsigned short)(u >> 16);
}

__device__ __forceinline__ float sigmoidf(float x) {
    return 1.0f / (1.0f + __expf(-x));
}

// Defeat LICM: make a pointer look loop-variant so weight loads stay in-loop
// (L2 hits) instead of being hoisted across 2048 iterations and spilled.
__device__ __forceinline__ const unsigned short* launder(const unsigned short* p) {
    unsigned long long v = (unsigned long long)p;
    asm volatile("" : "+v"(v));
    return (const unsigned short*)v;
}

// A fragment (16-bit 16x32 layout): lane holds K=[k0..k0+15] contiguous
__device__ __forceinline__ v16bf ldsA(const unsigned short* p) {
    u16x8 lo = *(const u16x8*)(p);
    u16x8 hi = *(const u16x8*)(p + 8);
    u16x16 a = __builtin_shufflevector(lo, hi, 0,1,2,3,4,5,6,7,8,9,10,11,12,13,14,15);
    return __builtin_bit_cast(v16bf, a);
}

// B fragment: 16 contiguous bf16, loaded via address_space(1) so the backend
// emits global_load_b128 (LOADcnt) even when the base pointer was laundered.
__device__ __forceinline__ v16bf ldgB(const unsigned short* p) {
    u16x16 b = *(gbl_u16x16_p)(unsigned long long)p;
    return __builtin_bit_cast(v16bf, b);
}

__device__ __forceinline__ v8f wmma_bf16(v16bf a, v16bf b, v8f c) {
    return __builtin_amdgcn_wmma_f32_16x16x32_bf16(
        /*neg_a=*/false, a, /*neg_b=*/false, b,
        /*c_mod=*/(short)0, c, /*reuse_a=*/false, /*reuse_b=*/false);
}

// ---------------------------------------------------------------------------
// Prelude: pack W/U into N-major bf16:  w1t[n][k] (k<128: W1, else U1),
// w2t[n][k] (k<256: W2, else U2), wft[n][k] = Wf[k][n].
// ---------------------------------------------------------------------------
__global__ void pack_weights_kernel(const float* __restrict__ W1,
                                    const float* __restrict__ U1,
                                    const float* __restrict__ W2,
                                    const float* __restrict__ U2,
                                    const float* __restrict__ Wf,
                                    unsigned short* __restrict__ w1t,
                                    unsigned short* __restrict__ w2t,
                                    unsigned short* __restrict__ wft) {
    const int tot1 = NG_ * K1_;
    const int tot2 = NG_ * K2_;
    const int tot3 = O_ * H_;
    int idx = blockIdx.x * blockDim.x + threadIdx.x;
    if (idx < tot1) {
        int n = idx / K1_, k = idx % K1_;
        float v = (k < D_) ? W1[k * NG_ + n] : U1[(k - D_) * NG_ + n];
        w1t[idx] = f2bf(v);
    } else if (idx < tot1 + tot2) {
        int j = idx - tot1;
        int n = j / K2_, k = j % K2_;
        float v = (k < H_) ? W2[k * NG_ + n] : U2[(k - H_) * NG_ + n];
        w2t[j] = f2bf(v);
    } else if (idx < tot1 + tot2 + tot3) {
        int j = idx - tot1 - tot2;
        int n = j / H_, k = j % H_;
        wft[j] = f2bf(Wf[k * O_ + n]);
    }
}

// ---------------------------------------------------------------------------
// Main fused kernel: grid = 4 blocks (batch blocks), 1024 threads each.
// ---------------------------------------------------------------------------
extern "C" __global__ void __launch_bounds__(1024, 1)
lstm_fused_kernel(const float* __restrict__ x,
                  const float* __restrict__ b1,
                  const float* __restrict__ b2,
                  const float* __restrict__ bfin,
                  const unsigned short* __restrict__ w1t,
                  const unsigned short* __restrict__ w2t,
                  const unsigned short* __restrict__ wft,
                  float* __restrict__ out) {
    extern __shared__ unsigned char smem[];
    float*          Z  = (float*)smem;                           // 16 x 1024 f32
    unsigned short* XA = (unsigned short*)(smem + 65536);        // 16 x 136 bf16
    unsigned short* H1 = XA + 16 * XA_STRIDE;                    // 16 x 264 bf16
    unsigned short* H2 = H1 + 16 * H_STRIDE;                     // 16 x 264 bf16

    const int tid  = threadIdx.x;
    const int wave = tid >> 5;
    const int lane = tid & 31;
    const int l15  = lane & 15;
    const int hiH  = lane >> 4;
    const int m    = blockIdx.x;              // batch rows [16m, 16m+16)

    const int aRow = l15;
    const int aOff = hiH * 16;                // A-fragment K sub-offset
    const int bOff = hiH * 16;                // B-fragment K sub-offset

    // zero h state
    for (int i = tid; i < 16 * H_STRIDE; i += 1024) { H1[i] = 0; H2[i] = 0; }

    float c1[4] = {0.f, 0.f, 0.f, 0.f};
    float c2[4] = {0.f, 0.f, 0.f, 0.f};

    const int nt0 = 2 * wave;                 // this wave's two N-tiles
    const float bias1_0 = b1[nt0 * 16 + l15];
    const float bias1_1 = b1[(nt0 + 1) * 16 + l15];
    const float bias2_0 = b2[nt0 * 16 + l15];
    const float bias2_1 = b2[(nt0 + 1) * 16 + l15];

    const unsigned short* pB1_0 = w1t + (size_t)(nt0 * 16 + l15) * K1_;
    const unsigned short* pB1_1 = w1t + (size_t)((nt0 + 1) * 16 + l15) * K1_;
    const unsigned short* pB2_0 = w2t + (size_t)(nt0 * 16 + l15) * K2_;
    const unsigned short* pB2_1 = w2t + (size_t)((nt0 + 1) * 16 + l15) * K2_;

    // per-thread x staging coordinates
    const int r0 = tid >> 7,          d0 = tid & 127;
    const int r1 = (tid + 1024) >> 7, d1 = tid & 127;  // i1 = tid+1024 -> same d
    const float* xr0 = x + (size_t)(16 * m + r0) * (T_ * D_) + d0;
    const float* xr1 = x + (size_t)(16 * m + r1) * (T_ * D_) + d1;

    __syncthreads();

    for (int t = 0; t < T_; ++t) {
        // ---- stage x_t (16 x 128 f32 -> bf16 LDS) + prefetch x_{t+2} ------
        {
            float v0 = xr0[(size_t)t * D_];
            float v1 = xr1[(size_t)t * D_];
            if (t + 2 < T_) {
                __builtin_prefetch(xr0 + (size_t)(t + 2) * D_, 0, 3);
                __builtin_prefetch(xr1 + (size_t)(t + 2) * D_, 0, 3);
            }
            XA[r0 * XA_STRIDE + d0] = f2bf(v0);
            XA[r1 * XA_STRIDE + d1] = f2bf(v1);
        }
        __syncthreads();

        // ---- layer-1 GEMM: z = [x_t | h1] @ [W1;U1] + b1 ------------------
        {
            const unsigned short* qB0 = launder(pB1_0);
            const unsigned short* qB1 = launder(pB1_1);
            v8f acc0, acc1;
#pragma unroll
            for (int r = 0; r < 8; ++r) { acc0[r] = bias1_0; acc1[r] = bias1_1; }
#pragma unroll 2
            for (int kc = 0; kc < 4; ++kc) {          // x part, K 0..127
                v16bf a  = ldsA(XA + aRow * XA_STRIDE + kc * 32 + aOff);
                v16bf bA = ldgB(qB0 + kc * 32 + bOff);
                v16bf bB = ldgB(qB1 + kc * 32 + bOff);
                acc0 = wmma_bf16(a, bA, acc0);
                acc1 = wmma_bf16(a, bB, acc1);
            }
#pragma unroll 2
            for (int kc = 0; kc < 8; ++kc) {          // h1 part, K 128..383
                v16bf a  = ldsA(H1 + aRow * H_STRIDE + kc * 32 + aOff);
                v16bf bA = ldgB(qB0 + 128 + kc * 32 + bOff);
                v16bf bB = ldgB(qB1 + 128 + kc * 32 + bOff);
                acc0 = wmma_bf16(a, bA, acc0);
                acc1 = wmma_bf16(a, bB, acc1);
            }
#pragma unroll
            for (int r = 0; r < 8; ++r) {
                Z[(r + 8 * hiH) * NG_ + nt0 * 16 + l15]       = acc0[r];
                Z[(r + 8 * hiH) * NG_ + (nt0 + 1) * 16 + l15] = acc1[r];
            }
        }
        __syncthreads();

        // ---- layer-1 gates ------------------------------------------------
#pragma unroll
        for (int e = 0; e < 4; ++e) {
            int idx = e * 1024 + tid;
            int r = idx >> 8, j = idx & 255;
            float zi = Z[r * NG_ + j];
            float zf = Z[r * NG_ + 256 + j];
            float zg = Z[r * NG_ + 512 + j];
            float zo = Z[r * NG_ + 768 + j];
            float c = sigmoidf(zf) * c1[e] + sigmoidf(zi) * zg;
            c1[e] = c;
            H1[r * H_STRIDE + j] = f2bf(sigmoidf(zo) * c);
        }
        __syncthreads();

        // ---- layer-2 GEMM: z = [h1 | h2] @ [W2;U2] + b2 -------------------
        {
            const unsigned short* qB0 = launder(pB2_0);
            const unsigned short* qB1 = launder(pB2_1);
            v8f acc0, acc1;
#pragma unroll
            for (int r = 0; r < 8; ++r) { acc0[r] = bias2_0; acc1[r] = bias2_1; }
#pragma unroll 2
            for (int kc = 0; kc < 8; ++kc) {          // h1 part, K 0..255
                v16bf a  = ldsA(H1 + aRow * H_STRIDE + kc * 32 + aOff);
                v16bf bA = ldgB(qB0 + kc * 32 + bOff);
                v16bf bB = ldgB(qB1 + kc * 32 + bOff);
                acc0 = wmma_bf16(a, bA, acc0);
                acc1 = wmma_bf16(a, bB, acc1);
            }
#pragma unroll 2
            for (int kc = 0; kc < 8; ++kc) {          // h2 part, K 256..511
                v16bf a  = ldsA(H2 + aRow * H_STRIDE + kc * 32 + aOff);
                v16bf bA = ldgB(qB0 + 256 + kc * 32 + bOff);
                v16bf bB = ldgB(qB1 + 256 + kc * 32 + bOff);
                acc0 = wmma_bf16(a, bA, acc0);
                acc1 = wmma_bf16(a, bB, acc1);
            }
#pragma unroll
            for (int r = 0; r < 8; ++r) {
                Z[(r + 8 * hiH) * NG_ + nt0 * 16 + l15]       = acc0[r];
                Z[(r + 8 * hiH) * NG_ + (nt0 + 1) * 16 + l15] = acc1[r];
            }
        }
        __syncthreads();

        // ---- layer-2 gates ------------------------------------------------
#pragma unroll
        for (int e = 0; e < 4; ++e) {
            int idx = e * 1024 + tid;
            int r = idx >> 8, j = idx & 255;
            float zi = Z[r * NG_ + j];
            float zf = Z[r * NG_ + 256 + j];
            float zg = Z[r * NG_ + 512 + j];
            float zo = Z[r * NG_ + 768 + j];
            float c = sigmoidf(zf) * c2[e] + sigmoidf(zi) * zg;
            c2[e] = c;
            H2[r * H_STRIDE + j] = f2bf(sigmoidf(zo) * c);
        }
        __syncthreads();
    }

    // ---- final dense: out = h2_last @ Wf + bf (16 x 64 per block) ---------
    if (wave < 4) {
        v8f acc;
#pragma unroll
        for (int r = 0; r < 8; ++r) acc[r] = bfin[wave * 16 + l15];
        const unsigned short* pBf = wft + (size_t)(wave * 16 + l15) * H_;
#pragma unroll 2
        for (int kc = 0; kc < 8; ++kc) {
            v16bf a = ldsA(H2 + aRow * H_STRIDE + kc * 32 + aOff);
            v16bf b = ldgB(pBf + kc * 32 + bOff);
            acc = wmma_bf16(a, b, acc);
        }
#pragma unroll
        for (int r = 0; r < 8; ++r)
            out[(size_t)(16 * m + r + 8 * hiH) * O_ + wave * 16 + l15] = acc[r];
    }
}

// ---------------------------------------------------------------------------
extern "C" void kernel_launch(void* const* d_in, const int* in_sizes, int n_in,
                              void* d_out, int out_size, void* d_ws, size_t ws_size,
                              hipStream_t stream) {
    (void)in_sizes; (void)n_in; (void)out_size; (void)ws_size;
    const float* x  = (const float*)d_in[0];
    const float* W1 = (const float*)d_in[1];
    const float* U1 = (const float*)d_in[2];
    const float* b1 = (const float*)d_in[3];
    const float* W2 = (const float*)d_in[4];
    const float* U2 = (const float*)d_in[5];
    const float* b2 = (const float*)d_in[6];
    const float* Wf = (const float*)d_in[7];
    const float* bf = (const float*)d_in[8];

    unsigned short* w1t = (unsigned short*)d_ws;            // 1024*384 bf16
    unsigned short* w2t = w1t + (size_t)NG_ * K1_;          // 1024*512 bf16
    unsigned short* wft = w2t + (size_t)NG_ * K2_;          // 64*256  bf16

    const int total = NG_ * K1_ + NG_ * K2_ + O_ * H_;
    pack_weights_kernel<<<(total + 255) / 256, 256, 0, stream>>>(
        W1, U1, W2, U2, Wf, w1t, w2t, wft);

    (void)hipFuncSetAttribute((const void*)lstm_fused_kernel,
                              hipFuncAttributeMaxDynamicSharedMemorySize,
                              SMEM_BYTES);
    lstm_fused_kernel<<<dim3(4), dim3(1024), SMEM_BYTES, stream>>>(
        x, b1, b2, bf, w1t, w2t, wft, (float*)d_out);
}
